// DifferentialDropout_17575006175621
// MI455X (gfx1250) — matrix operand order repaired and checked by reading, not compiled, not run on hardware
//
#include <hip/hip_runtime.h>
#include <stdint.h>
#include <stddef.h>

// ---------------------------------------------------------------------------
// DifferentialDropout for MI455X (gfx1250, wave32).
// Gram matrix (corrcoef core) via v_wmma_f32_16x16x32_bf16 with hi/lo fp32
// split (3 WMMA products) and register double-buffered fragment loads,
// presence-bitmap replacement for the sort-based unique counts, streaming
// passes for everything else.
// ---------------------------------------------------------------------------

#define B_ROWS 256
#define F_COLS 65536

typedef __attribute__((ext_vector_type(16))) __bf16 bf16x16;
typedef __attribute__((ext_vector_type(8)))  float  floatx8;

union FragU {
    bf16x16 f;
    uint4   q[2];
};

__device__ __forceinline__ unsigned short f2bf(float f) {
    unsigned u = __builtin_bit_cast(unsigned, f);
    u += 0x7FFFu + ((u >> 16) & 1u);       // round-to-nearest-even
    return (unsigned short)(u >> 16);
}
__device__ __forceinline__ float bf2f(unsigned short h) {
    unsigned u = ((unsigned)h) << 16;
    return __builtin_bit_cast(float, u);
}
__device__ __forceinline__ void atomic_add_f32(float* p, float v) {
    __hip_atomic_fetch_add(p, v, __ATOMIC_RELAXED, __HIP_MEMORY_SCOPE_AGENT);
}

// ---------------------------------------------------------------------------
// Workspace layout (bytes)
// ---------------------------------------------------------------------------
static constexpr size_t OFF_XH  = 0;                                   // 32 MB bf16 hi
static constexpr size_t OFF_XL  = OFF_XH + (size_t)B_ROWS * F_COLS * 2;// 32 MB bf16 lo
static constexpr size_t OFF_G   = OFF_XL + (size_t)B_ROWS * F_COLS * 2;// 256 KB Gram f32
static constexpr size_t OFF_GBM = OFF_G   + (size_t)B_ROWS * B_ROWS * 4;// 32 B global bitmap
static constexpr size_t OFF_SM2 = OFF_GBM + 32;                        // 4 B (+pad) sum(colmean^2)
static constexpr size_t OFF_RBM = OFF_SM2 + 32;                        // 8 KB row bitmaps
static constexpr size_t OFF_CM  = OFF_RBM + (size_t)B_ROWS * 8 * 4;    // 256 KB col means
static constexpr size_t OFF_RS  = OFF_CM  + (size_t)F_COLS * 4;        // 1 KB row sums
static constexpr size_t OFF_RQ  = OFF_RS  + B_ROWS * 4;                // 1 KB row sumsq
static constexpr size_t OFF_RD  = OFF_RQ  + B_ROWS * 4;                // 1 KB row dot colmean
static constexpr size_t OFF_P   = OFF_RD  + B_ROWS * 4;                // 1 KB p[]
static constexpr size_t OFF_SC  = OFF_P   + B_ROWS * 4;                // 4 B scale
static constexpr size_t ZERO_WORDS = (OFF_CM - OFF_G) / 4;             // G + bitmaps + Sm2

// ---------------------------------------------------------------------------
__global__ void zero_kernel(unsigned* __restrict__ w, int n) {
    int i = blockIdx.x * blockDim.x + threadIdx.x;
    if (i < n) w[i] = 0u;
}

// fp32 -> (bf16 hi, bf16 lo residual), 8 elems/thread
__global__ void split_kernel(const float* __restrict__ x,
                             unsigned short* __restrict__ xh,
                             unsigned short* __restrict__ xl) {
    long long t = (long long)blockIdx.x * blockDim.x + threadIdx.x;
    long long base = t * 8;
    float4 a = *(const float4*)(x + base);
    float4 b = *(const float4*)(x + base + 4);
    float v[8] = {a.x, a.y, a.z, a.w, b.x, b.y, b.z, b.w};
    unsigned short h[8], l[8];
#pragma unroll
    for (int i = 0; i < 8; i++) {
        h[i] = f2bf(v[i]);
        l[i] = f2bf(v[i] - bf2f(h[i]));
    }
    uint4 ho = {(unsigned)h[0] | ((unsigned)h[1] << 16),
                (unsigned)h[2] | ((unsigned)h[3] << 16),
                (unsigned)h[4] | ((unsigned)h[5] << 16),
                (unsigned)h[6] | ((unsigned)h[7] << 16)};
    uint4 lo = {(unsigned)l[0] | ((unsigned)l[1] << 16),
                (unsigned)l[2] | ((unsigned)l[3] << 16),
                (unsigned)l[4] | ((unsigned)l[5] << 16),
                (unsigned)l[6] | ((unsigned)l[7] << 16)};
    *(uint4*)(xh + base) = ho;
    *(uint4*)(xl + base) = lo;
}

// column means + sum of squared col means
__global__ void colmean_kernel(const float* __restrict__ x,
                               float* __restrict__ cm,
                               float* __restrict__ sm2) {
    int j = blockIdx.x * blockDim.x + threadIdx.x;
    float s = 0.f;
    for (int i = 0; i < B_ROWS; i++) s += x[(size_t)i * F_COLS + j];
    float m = s * (1.0f / B_ROWS);
    cm[j] = m;
    __shared__ float red[256];
    red[threadIdx.x] = m * m;
    __syncthreads();
    for (int o = 128; o > 0; o >>= 1) {
        if (threadIdx.x < o) red[threadIdx.x] += red[threadIdx.x + o];
        __syncthreads();
    }
    if (threadIdx.x == 0) atomic_add_f32(sm2, red[0]);
}

// per-row: sum, sumsq, dot(colmean), rounded-value presence bitmap
__global__ void rowstats_kernel(const float* __restrict__ x,
                                const float* __restrict__ cm,
                                float* __restrict__ rs, float* __restrict__ rq,
                                float* __restrict__ rd,
                                unsigned* __restrict__ rbm,
                                unsigned* __restrict__ gbm) {
    int row = blockIdx.x, tid = threadIdx.x;
    __shared__ unsigned bm[8];
    __shared__ float red[256];
    if (tid < 8) bm[tid] = 0u;
    __syncthreads();
    const float* xr = x + (size_t)row * F_COLS;
    float s1 = 0.f, s2 = 0.f, sd = 0.f;
    for (int j = tid; j < F_COLS; j += 256) {
        float v = xr[j];
        s1 += v; s2 += v * v; sd += v * cm[j];
        int b = (int)rintf(v);
        b = b < -128 ? -128 : (b > 127 ? 127 : b);
        unsigned bin = (unsigned)(b + 128);
        atomicOr(&bm[bin >> 5], 1u << (bin & 31));
    }
    red[tid] = s1; __syncthreads();
    for (int o = 128; o > 0; o >>= 1) { if (tid < o) red[tid] += red[tid + o]; __syncthreads(); }
    if (tid == 0) rs[row] = red[0];
    __syncthreads();
    red[tid] = s2; __syncthreads();
    for (int o = 128; o > 0; o >>= 1) { if (tid < o) red[tid] += red[tid + o]; __syncthreads(); }
    if (tid == 0) rq[row] = red[0];
    __syncthreads();
    red[tid] = sd; __syncthreads();
    for (int o = 128; o > 0; o >>= 1) { if (tid < o) red[tid] += red[tid + o]; __syncthreads(); }
    if (tid == 0) rd[row] = red[0];
    __syncthreads();
    if (tid < 8) {
        rbm[row * 8 + tid] = bm[tid];
        atomicOr(&gbm[tid], bm[tid]);
    }
}

// ---------------------------------------------------------------------------
// Gram matrix G = X * X^T via bf16 hi/lo split WMMA (hh + hl + lh).
// Upper-triangular tiles (rt<=ct), K split into 16 segments, mirror+sum via
// f32 atomics. One wave per (tile, segment): 136*16 = 2176 waves.
// Fragment loads are register double-buffered (ping-pong) so each WMMA group
// only waits for loads issued one iteration earlier while the next set stays
// in flight (partial s_wait_loadcnt instead of draining to 0).
// ---------------------------------------------------------------------------
__device__ __forceinline__ void load_frags(const unsigned short* __restrict__ ah0,
                                           const unsigned short* __restrict__ al0,
                                           const unsigned short* __restrict__ bh0,
                                           const unsigned short* __restrict__ bl0,
                                           int k, int half,
                                           FragU& Ah, FragU& Al, FragU& Bh, FragU& Bl) {
    // A-frag (16x32 bf16): lanes 0-15 rows M=l16 K 0..7 & 16..23,
    // lanes 16-31 same rows K 8..15 & 24..31 -> two 16B chunks per lane.
    size_t ao = (size_t)k + (size_t)half * 8;
    Ah.q[0] = *(const uint4*)(ah0 + ao);
    Ah.q[1] = *(const uint4*)(ah0 + ao + 16);
    Al.q[0] = *(const uint4*)(al0 + ao);
    Al.q[1] = *(const uint4*)(al0 + ao + 16);
    // B-frag (32x16 bf16): lane n holds column n (= row of X), K 0..7 & 16..23
    // (lanes 0-15) or K 8..15 & 24..31 (lanes 16-31) -> two 16B chunks per lane.
    size_t bo = (size_t)k + (size_t)half * 8;
    Bh.q[0] = *(const uint4*)(bh0 + bo);
    Bh.q[1] = *(const uint4*)(bh0 + bo + 16);
    Bl.q[0] = *(const uint4*)(bl0 + bo);
    Bl.q[1] = *(const uint4*)(bl0 + bo + 16);
}

__global__ __launch_bounds__(256)
void gram_kernel(const unsigned short* __restrict__ xh,
                 const unsigned short* __restrict__ xl,
                 float* __restrict__ G) {
    const int SEG = 16, SEGLEN = F_COLS / SEG;   // 4096 -> 128 chunks of 32
    int wid  = (int)((blockIdx.x * blockDim.x + threadIdx.x) >> 5);
    int lane = threadIdx.x & 31;
    int tile = wid / SEG, seg = wid % SEG;

    // tile -> (rt, ct) with rt <= ct over a 16x16 tile grid
    int rt = 0, t = tile;
    while (t >= 16 - rt) { t -= 16 - rt; rt++; }
    int ct = rt + t;

    int half = lane >> 4, l16 = lane & 15;
    const unsigned short* ah0 = xh + (size_t)(rt * 16 + l16) * F_COLS;
    const unsigned short* al0 = xl + (size_t)(rt * 16 + l16) * F_COLS;
    const unsigned short* bh0 = xh + (size_t)(ct * 16 + l16) * F_COLS;
    const unsigned short* bl0 = xl + (size_t)(ct * 16 + l16) * F_COLS;

    floatx8 chh = {}, chl = {}, clh = {};
    int k0 = seg * SEGLEN;

    FragU A0h, A0l, B0h, B0l;   // set 0
    FragU A1h, A1l, B1h, B1l;   // set 1

    load_frags(ah0, al0, bh0, bl0, k0, half, A0h, A0l, B0h, B0l);

    int k = k0;
#pragma unroll 2
    for (int it = 0; it < SEGLEN / 64; ++it) {
        // issue loads for set1 before consuming set0
        load_frags(ah0, al0, bh0, bl0, k + 32, half, A1h, A1l, B1h, B1l);
        chh = __builtin_amdgcn_wmma_f32_16x16x32_bf16(false, A0h.f, false, B0h.f,
                                                      (short)0, chh, false, false);
        chl = __builtin_amdgcn_wmma_f32_16x16x32_bf16(false, A0h.f, false, B0l.f,
                                                      (short)0, chl, false, false);
        clh = __builtin_amdgcn_wmma_f32_16x16x32_bf16(false, A0l.f, false, B0h.f,
                                                      (short)0, clh, false, false);
        // issue loads for next set0 before consuming set1 (last iteration
        // prefetches <=128B past the segment: still inside d_ws, discarded)
        load_frags(ah0, al0, bh0, bl0, k + 64, half, A0h, A0l, B0h, B0l);
        chh = __builtin_amdgcn_wmma_f32_16x16x32_bf16(false, A1h.f, false, B1h.f,
                                                      (short)0, chh, false, false);
        chl = __builtin_amdgcn_wmma_f32_16x16x32_bf16(false, A1h.f, false, B1l.f,
                                                      (short)0, chl, false, false);
        clh = __builtin_amdgcn_wmma_f32_16x16x32_bf16(false, A1l.f, false, B1h.f,
                                                      (short)0, clh, false, false);
        k += 64;
    }
    floatx8 g = chh + chl + clh;

    // C/D layout: VGPR r, lanes 0-15 -> (M=r, N=lane); lanes 16-31 -> (M=8+r)
#pragma unroll
    for (int r = 0; r < 8; r++) {
        int row = rt * 16 + half * 8 + r;
        int col = ct * 16 + l16;
        float v = g[r];
        atomic_add_f32(&G[row * B_ROWS + col], v);
        if (rt != ct) atomic_add_f32(&G[col * B_ROWS + row], v);
    }
}

// ---------------------------------------------------------------------------
// Single-block finalize: corr factors, mse factors, unique factors -> p[]
// ---------------------------------------------------------------------------
__global__ void finalize_kernel(const float* __restrict__ G,
                                const float* __restrict__ rs,
                                const float* __restrict__ rq,
                                const float* __restrict__ rd,
                                const unsigned* __restrict__ rbm,
                                const unsigned* __restrict__ gbm,
                                const float* __restrict__ sm2p,
                                const int* __restrict__ epochp,
                                float* __restrict__ p,
                                float* __restrict__ scalep) {
    int i = threadIdx.x;
    const float Ff = (float)F_COLS;
    __shared__ float mu[256], rsq[256], red[256];
    __shared__ float tm_sh;
    float mui = rs[i] / Ff;
    mu[i] = mui;
    float dii = G[i * B_ROWS + i] - Ff * mui * mui;
    float ri = rsqrtf(fmaxf(dii, 1e-30f));
    rsq[i] = ri;
    __syncthreads();
    float acc = 0.f;
    for (int k = 0; k < B_ROWS; k++) {
        float gc = G[i * B_ROWS + k] - Ff * mui * mu[k];
        float c = gc * ri * rsq[k];
        c = fminf(fmaxf(c, -1.f), 1.f);          // corrcoef clips to [-1,1]
        acc += fabsf(c);
    }
    float f1 = acc / (float)B_ROWS;

    float mse = (rq[i] - 2.f * rd[i] + *sm2p) / Ff;
    red[i] = mse; __syncthreads();
    for (int o = 128; o > 0; o >>= 1) { if (i < o) red[i] += red[i + o]; __syncthreads(); }
    if (i == 0) tm_sh = red[0];
    __syncthreads();
    float f2 = mse / tm_sh;

    int ru = 0, tu = 0;
#pragma unroll
    for (int w = 0; w < 8; w++) { ru += __popc(rbm[i * 8 + w]); tu += __popc(gbm[w]); }
    float f3 = (float)ru / (float)tu;

    double ed = (double)epochp[0];
    double t4 = ed * sin(ed) * cos(ed);
    float f4 = (float)(ed + t4 * t4);

    float pi = 1.f - (f1 + (1.f - f2) + (1.f - f3)) / (3.f * f4);
    p[i] = pi;
    if (i == B_ROWS - 1) *scalep = 1.f / (1.f - pi);
}

// out = (noise > p[row]) * x * (1/(1-p_last)), float4 streams
__global__ void output_kernel(const float* __restrict__ x,
                              const float* __restrict__ noise,
                              const float* __restrict__ p,
                              const float* __restrict__ scalep,
                              float* __restrict__ out) {
    long long t = (long long)blockIdx.x * blockDim.x + threadIdx.x;
    long long base = t * 4;
    int row = (int)(base >> 16);                  // F = 65536
    float pr = p[row];
    float s  = *scalep;
    float4 xv = *(const float4*)(x + base);
    float4 nv = *(const float4*)(noise + base);
    float4 o;
    o.x = nv.x > pr ? xv.x * s : 0.f;
    o.y = nv.y > pr ? xv.y * s : 0.f;
    o.z = nv.z > pr ? xv.z * s : 0.f;
    o.w = nv.w > pr ? xv.w * s : 0.f;
    *(float4*)(out + base) = o;
}

// ---------------------------------------------------------------------------
extern "C" void kernel_launch(void* const* d_in, const int* in_sizes, int n_in,
                              void* d_out, int out_size, void* d_ws, size_t ws_size,
                              hipStream_t stream) {
    (void)in_sizes; (void)n_in; (void)out_size; (void)ws_size;
    const float* x     = (const float*)d_in[0];
    const float* noise = (const float*)d_in[1];
    const int*   epoch = (const int*)d_in[2];
    float*       out   = (float*)d_out;

    char* ws = (char*)d_ws;
    unsigned short* xh = (unsigned short*)(ws + OFF_XH);
    unsigned short* xl = (unsigned short*)(ws + OFF_XL);
    float*    G    = (float*)(ws + OFF_G);
    unsigned* gbm  = (unsigned*)(ws + OFF_GBM);
    float*    sm2  = (float*)(ws + OFF_SM2);
    unsigned* rbm  = (unsigned*)(ws + OFF_RBM);
    float*    cm   = (float*)(ws + OFF_CM);
    float*    rs   = (float*)(ws + OFF_RS);
    float*    rq   = (float*)(ws + OFF_RQ);
    float*    rd   = (float*)(ws + OFF_RD);
    float*    p    = (float*)(ws + OFF_P);
    float*    sc   = (float*)(ws + OFF_SC);

    const long long n = (long long)B_ROWS * F_COLS;   // 16,777,216

    zero_kernel<<<(int)((ZERO_WORDS + 255) / 256), 256, 0, stream>>>(
        (unsigned*)(ws + OFF_G), (int)ZERO_WORDS);

    split_kernel<<<(int)(n / 8 / 256), 256, 0, stream>>>(x, xh, xl);

    colmean_kernel<<<F_COLS / 256, 256, 0, stream>>>(x, cm, sm2);

    rowstats_kernel<<<B_ROWS, 256, 0, stream>>>(x, cm, rs, rq, rd, rbm, gbm);

    // 136 upper-tri tiles * 16 K-segments = 2176 waves = 272 blocks of 8 waves
    gram_kernel<<<272, 256, 0, stream>>>(xh, xl, G);

    finalize_kernel<<<1, 256, 0, stream>>>(G, rs, rq, rd, rbm, gbm, sm2, epoch, p, sc);

    output_kernel<<<(int)(n / 4 / 256), 256, 0, stream>>>(x, noise, p, sc, out);
}
